// BahdanauAttention_51402168599064
// MI455X (gfx1250) — matrix operand order repaired and checked by reading, not compiled, not run on hardware
//
#include <hip/hip_runtime.h>
#include <stdint.h>
#include <stddef.h>

// Problem constants (from reference): B=32, T=2048, D=1024, N=1024
#define BSZ 32
#define TN  2048
#define DD  1024
#define NN  1024
#define MTILE 128
#define ROWB  2064                       // padded LDS row stride (1024 bf16 = 2048B, +16)
#define SCORE_OFF (MTILE * ROWB)         // per-WG row-score accumulator in LDS
#define LDS_BYTES (SCORE_OFF + MTILE * 4)   // 264704 B < 320 KB per-WG LDS on CDNA5

typedef __attribute__((ext_vector_type(16))) __bf16 v16bf;
typedef __attribute__((ext_vector_type(8)))  float  v8f;

union BF16Frag { uint4 u[2]; v16bf v; };

__device__ __forceinline__ uint16_t f32_to_bf16_rne(float f) {
  uint32_t u = __builtin_bit_cast(uint32_t, f);
  uint32_t r = u + 0x7FFFu + ((u >> 16) & 1u);   // round-to-nearest-even
  return (uint16_t)(r >> 16);
}

// Hardware tanh (CDNA5 TRANS op, co-executes with WMMA).
__device__ __forceinline__ float htanh(float x) {
#if __has_builtin(__builtin_amdgcn_tanhf)
  return __builtin_amdgcn_tanhf(x);
#elif __has_builtin(__builtin_amdgcn_tanh_f32)
  return __builtin_amdgcn_tanh_f32(x);
#else
  float r;
  // v_nop covers the documented TRANS-result hazard (1 indep op before use).
  asm volatile("v_tanh_f32 %0, %1\n\tv_nop" : "=v"(r) : "v"(x));
  return r;
#endif
}

// ---------------------------------------------------------------------------
// Kernel 0: repack W2 (f32 [D,N]) -> bf16 WMMA-B-fragment-major layout.
// Fragment block (kt, nt) = K rows [kt*32, +32) x N cols [nt*16, +16).
// Lane (B operand, this code's contiguous k order, wave32): n = lane%16, g = lane/16; elements e=0..15 hold
// K = g*16 + e. Stored contiguously: 32 lanes x 32 bytes = 1 KB per fragment.
// Fragment f = kt*64 + nt lives at byte offset f*1024 + lane*32.
// ---------------------------------------------------------------------------
__global__ void pack_w2(const float* __restrict__ W2, uint16_t* __restrict__ pk) {
  int f    = blockIdx.x * blockDim.x + threadIdx.x;   // 65536 fragment-lanes
  int lane = f & 31;
  int nt   = (f >> 5) & 63;
  int kt   = f >> 11;
  int g = lane >> 4, n = lane & 15;
  uint16_t tmp[16];
#pragma unroll
  for (int e = 0; e < 16; ++e) {
    int k = kt * 32 + g * 16 + e;
    tmp[e] = f32_to_bf16_rne(W2[(size_t)k * NN + nt * 16 + n]);
  }
  uint4* dst = (uint4*)(pk + (size_t)f * 16);
  dst[0] = *(uint4*)&tmp[0];
  dst[1] = *(uint4*)&tmp[8];
}

// ---------------------------------------------------------------------------
// Kernel 1: qpb[b][n] = query[b]@W1[:,n] + b1[n] + b2[n]   (b2 folded in)
// ---------------------------------------------------------------------------
__global__ void qproj(const float* __restrict__ q, const float* __restrict__ W1,
                      const float* __restrict__ b1, const float* __restrict__ b2,
                      float* __restrict__ qpb) {
  int n = blockIdx.x * blockDim.x + threadIdx.x;
  int b = blockIdx.y;
  float acc = b1[n] + b2[n];
  const float* qr = q + (size_t)b * DD;
#pragma unroll 4
  for (int k = 0; k < DD; ++k) acc += qr[k] * W1[(size_t)k * NN + n];
  qpb[(size_t)b * NN + n] = acc;
}

// ---------------------------------------------------------------------------
// Kernel 2 (hot): fused  score[b,t] = sum_n tanh(qpb[b,n] + values[b,t,:]@W2[:,n]) * V[n]
// 512 WGs x 8 waves. WG stages a 128x1024 tile of values as bf16 in LDS.
// Waves partition N (128 exclusive cols each). Per (npass, mpass):
//   4 Mtile x 4 Ntile f32 accumulators (128 VGPRs); kt(32): load 4 B frags +
//   4 A frags into distinct register arrays, then 16 back-to-back WMMAs.
// A reuse x4, B reuse x4; W2 fetched only 2x per WG from L2-resident pack.
// Partial per-row scores (disjoint n-subsets per wave) combine via ds_add_f32.
// ---------------------------------------------------------------------------
__global__ __launch_bounds__(256, 1) void score_kernel(
    const float* __restrict__ values, const uint16_t* __restrict__ w2pk,
    const float* __restrict__ qpb, const float* __restrict__ Vv,
    float* __restrict__ score) {
  extern __shared__ char smem[];
  float* lds_score = (float*)(smem + SCORE_OFF);
  const int tid  = threadIdx.x;
  const int row0 = blockIdx.x * MTILE;     // flattened (b,t) row; tile stays in one batch
  const int b    = row0 / TN;

  // ---- stage values[row0 .. +127][0..1023] f32 -> bf16 LDS (padded rows)
  const float4* src = (const float4*)(values + (size_t)row0 * DD);
  for (int idx = tid; idx < MTILE * (DD / 4); idx += 256) {
    int r  = idx >> 8;            // DD/4 == 256 float4 per row
    int c4 = idx & 255;
    float4 fv = src[(size_t)r * (DD / 4) + c4];
    uint16_t h[4] = { f32_to_bf16_rne(fv.x), f32_to_bf16_rne(fv.y),
                      f32_to_bf16_rne(fv.z), f32_to_bf16_rne(fv.w) };
    *(uint2*)(smem + r * ROWB + c4 * 8) = *(uint2*)h;
  }
  if (tid < MTILE) lds_score[tid] = 0.0f;
  __syncthreads();

  const int wave = tid >> 5, lane = tid & 31;
  const int g = lane >> 4, nin = lane & 15;
  // A-fragment (contiguous k order, same as the W2 pack): lane's row M = lane%16; K-halves by g.
  const char* abase = smem + nin * ROWB + g * 32;

#pragma unroll 1
  for (int npass = 0; npass < 2; ++npass) {
    const int n0 = wave * 128 + npass * 64;      // this wave's 4 n-tiles
    float qn[4], vn[4];
#pragma unroll
    for (int j = 0; j < 4; ++j) {
      qn[j] = qpb[(size_t)b * NN + n0 + j * 16 + nin];
      vn[j] = Vv[n0 + j * 16 + nin];
    }
    // B fragment base for this lane: frag (kt, nt0+j) at bbase + kt*65536 + j*1024
    const char* bbase = (const char*)w2pk + (size_t)(n0 >> 4) * 1024 + lane * 32;

#pragma unroll 1
    for (int mp = 0; mp < 2; ++mp) {
      // per-lane A base addresses for the 4 m-tiles of this mpass
      const char* am[4];
#pragma unroll
      for (int mt = 0; mt < 4; ++mt)
        am[mt] = abase + (mp * 64 + mt * 16) * ROWB;

      v8f acc[16];                               // [mt][j]
#pragma unroll
      for (int i = 0; i < 16; ++i) acc[i] = (v8f){};

#pragma unroll 2
      for (int kt = 0; kt < DD / 32; ++kt) {
        BF16Frag bf[4];
#pragma unroll
        for (int j = 0; j < 4; ++j) {
          bf[j].u[0] = *(const uint4*)(bbase + (size_t)kt * 65536 + j * 1024);
          bf[j].u[1] = *(const uint4*)(bbase + (size_t)kt * 65536 + j * 1024 + 16);
        }
        BF16Frag a[4];
#pragma unroll
        for (int mt = 0; mt < 4; ++mt) {
          a[mt].u[0] = *(const uint4*)(am[mt] + kt * 64);
          a[mt].u[1] = *(const uint4*)(am[mt] + kt * 64 + 16);
        }
#pragma unroll
        for (int mt = 0; mt < 4; ++mt) {
#pragma unroll
          for (int j = 0; j < 4; ++j)
            acc[mt * 4 + j] = __builtin_amdgcn_wmma_f32_16x16x32_bf16(
                false, a[mt].v, false, bf[j].v, (short)0, acc[mt * 4 + j], false, false);
        }
      }

      // ---- epilogue: tanh + V-weighted reduce over this wave's n-subset
#pragma unroll
      for (int mt = 0; mt < 4; ++mt) {
        float sacc[8];
#pragma unroll
        for (int r = 0; r < 8; ++r) {
          float s = 0.0f;
#pragma unroll
          for (int j = 0; j < 4; ++j)
            s += htanh(acc[mt * 4 + j][r] + qn[j]) * vn[j];
          sacc[r] = s;
        }
        // reduce across the 16 lanes of the half-wave (same m = r + 8g)
#pragma unroll
        for (int r = 0; r < 8; ++r) {
          float v = sacc[r];
          for (int off = 1; off < 16; off <<= 1) v += __shfl_xor(v, off, 16);
          sacc[r] = v;
        }
        if (nin == 0) {
#pragma unroll
          for (int r = 0; r < 8; ++r)
            atomicAdd(&lds_score[mp * 64 + mt * 16 + g * 8 + r], sacc[r]);  // ds_add_f32
        }
      }
    }
  }

  __syncthreads();
  if (tid < MTILE) score[row0 + tid] = lds_score[tid];
}

// ---------------------------------------------------------------------------
// Kernel 3: softmax over T per batch. (bV is softmax-invariant -> dropped.)
// ---------------------------------------------------------------------------
__global__ void softmax_k(const float* __restrict__ score,
                          float* __restrict__ attn_ws, float* __restrict__ attn_out) {
  int b = blockIdx.x, tid = threadIdx.x;
  __shared__ float red[256];
  __shared__ float ex[TN];
  const float* s = score + (size_t)b * TN;
  float m = -3.4e38f;
  for (int i = tid; i < TN; i += 256) m = fmaxf(m, s[i]);
  red[tid] = m; __syncthreads();
  for (int o = 128; o > 0; o >>= 1) {
    if (tid < o) red[tid] = fmaxf(red[tid], red[tid + o]);
    __syncthreads();
  }
  m = red[0]; __syncthreads();
  float sum = 0.0f;
  for (int i = tid; i < TN; i += 256) { float e = __expf(s[i] - m); ex[i] = e; sum += e; }
  red[tid] = sum; __syncthreads();
  for (int o = 128; o > 0; o >>= 1) {
    if (tid < o) red[tid] += red[tid + o];
    __syncthreads();
  }
  float inv = 1.0f / red[0];
  for (int i = tid; i < TN; i += 256) {
    float a = ex[i] * inv;
    attn_ws[(size_t)b * TN + i]  = a;
    attn_out[(size_t)b * TN + i] = a;
  }
}

// ---------------------------------------------------------------------------
// Kernel 4: context[b,d] = sum_t attn[b,t] * values[b,t,d]  (memory bound)
// ---------------------------------------------------------------------------
__global__ void context_k(const float* __restrict__ values,
                          const float* __restrict__ attn, float* __restrict__ out) {
  int b = blockIdx.y;
  int d = blockIdx.x * blockDim.x + threadIdx.x;
  __shared__ float aw[TN];
  for (int i = threadIdx.x; i < TN; i += 256) aw[i] = attn[(size_t)b * TN + i];
  __syncthreads();
  float acc = 0.0f;
  const float* vb = values + (size_t)b * TN * DD + d;
#pragma unroll 8
  for (int t = 0; t < TN; ++t) acc += aw[t] * vb[(size_t)t * DD];
  out[(size_t)b * DD + d] = acc;
}

// ---------------------------------------------------------------------------
extern "C" void kernel_launch(void* const* d_in, const int* in_sizes, int n_in,
                              void* d_out, int out_size, void* d_ws, size_t ws_size,
                              hipStream_t stream) {
  (void)in_sizes; (void)n_in; (void)out_size; (void)ws_size;
  const float* query  = (const float*)d_in[0];
  const float* values = (const float*)d_in[1];
  const float* W1     = (const float*)d_in[2];
  const float* b1     = (const float*)d_in[3];
  const float* W2     = (const float*)d_in[4];
  const float* b2     = (const float*)d_in[5];
  const float* Vv     = (const float*)d_in[6];
  // d_in[7] = bV: softmax-invariant, unused.

  float* out_ctx  = (float*)d_out;                 // [B, D]
  float* out_attn = (float*)d_out + BSZ * DD;      // [B, T, 1]

  char* ws = (char*)d_ws;                          // needs ~2.7 MB
  uint16_t* w2pk = (uint16_t*)ws;                                  // 2 MB
  float* qpb   = (float*)(ws + (2u << 20));                        // 128 KB
  float* score = (float*)(ws + (2u << 20) + (128u << 10));         // 256 KB
  float* attn  = (float*)(ws + (2u << 20) + (384u << 10));         // 256 KB

  pack_w2<<<256, 256, 0, stream>>>(W2, w2pk);
  qproj<<<dim3(NN / 256, BSZ), 256, 0, stream>>>(query, W1, b1, b2, qpb);
  score_kernel<<<(BSZ * TN) / MTILE, 256, LDS_BYTES, stream>>>(values, w2pk, qpb, Vv, score);
  softmax_k<<<BSZ, 256, 0, stream>>>(score, attn, out_attn);
  context_k<<<dim3(DD / 256, BSZ), 256, 0, stream>>>(values, attn, out_ctx);
}
